// MultimodalFusion_54400055771546
// MI455X (gfx1250) — matrix operand order repaired and checked by reading, not compiled, not run on hardware
//
#include <hip/hip_runtime.h>

// ---------------------------------------------------------------------------
// MultimodalFusion for gfx1250 (MI455X).
//  - All dense projections via v_wmma_f32_16x16x32_bf16 (wave32).
//  - A/B operands pre-swizzled in global memory into fragment-contiguous
//    order (ISA 7.12.2 VGPR layouts) so fragments are b128 loads.
//  - B slabs double-buffered through LDS with async global->LDS loads
//    (GLOBAL_LOAD_ASYNC_TO_LDS_B128 + s_wait_asynccnt) when available.
//  - Selective scan kept register-resident, one thread per channel.
// ---------------------------------------------------------------------------

#define ROWS   16384      // B * L tokens
#define LTOK   4096
#define BATCH  4
#define DM     96
#define NPAD   48         // padded width of x_dbl (6 + 16 + 16 -> 48)
#define DSTATE 16
#define BSLAB  4096       // u16 per (padded) B k-slab: 8KB, 512 b128 chunks

#define AS1 __attribute__((address_space(1)))
#define AS3 __attribute__((address_space(3)))

#if __has_builtin(__builtin_amdgcn_global_load_async_to_lds_b128) && \
    __has_builtin(__builtin_amdgcn_s_wait_asynccnt)
#define USE_ASYNC_COPY 1
#else
#define USE_ASYNC_COPY 0
#endif

typedef __attribute__((ext_vector_type(16))) __bf16 v16bf;
typedef __attribute__((ext_vector_type(8)))  float  v8f;
typedef int v4i_g __attribute__((vector_size(16)));   // matches builtin param

__device__ __forceinline__ unsigned short f2bfu(float f) {
  unsigned u = __float_as_uint(f);
  u += 0x7FFFu + ((u >> 16) & 1u);           // RNE to bf16
  return (unsigned short)(u >> 16);
}
__device__ __forceinline__ float silu_f(float v) {
  return v * (1.0f / (1.0f + __expf(-v)));
}

// ------------------------------- GEMM (WMMA) -------------------------------
// C[M, NT*16] = Asw x Bsw (+bias)(+SiLU).  256 threads = 8 waves; wave w owns
// rows 128*blk + 16*w .. +15.  K stepped in slabs of 32.
// Asw: fragment-contiguous: [((rowTile*KS + S)*32 + lane)*16 + j]
// Bsw: slab-major padded:   [S*BSLAB + (t*32 + lane)*16 + j]
template <int NT, int ACT>
__global__ void __launch_bounds__(256) gemm_bf16_wmma(
    const unsigned short* __restrict__ Asw, const unsigned short* __restrict__ Bsw,
    const float* __restrict__ bias, float* __restrict__ C, int K)
{
  __shared__ __attribute__((aligned(32))) unsigned short sB[2][BSLAB];
  const int lane = threadIdx.x & 31;
  const int wave = threadIdx.x >> 5;
  const int KS   = K >> 5;
  const long T   = (long)blockIdx.x * 8 + wave;   // 16-row tile index
  const int N    = NT * 16;

  v8f zero = {};
  v8f acc[NT];
#pragma unroll
  for (int t = 0; t < NT; ++t) acc[t] = zero;

  auto stageB = [&](int S, int buf) {
    const unsigned short* src = Bsw + (long)S * BSLAB;
    unsigned short* dst = &sB[buf][0];
    // exactly 2 x b128 per thread (512 chunks) -> constant wait immediates
#if USE_ASYNC_COPY
    __builtin_amdgcn_global_load_async_to_lds_b128(
        (AS1 v4i_g*)(src + threadIdx.x * 8),
        (AS3 v4i_g*)(dst + threadIdx.x * 8), 0, 0);
    __builtin_amdgcn_global_load_async_to_lds_b128(
        (AS1 v4i_g*)(src + (threadIdx.x + 256) * 8),
        (AS3 v4i_g*)(dst + (threadIdx.x + 256) * 8), 0, 0);
#else
    *(uint4*)(dst + threadIdx.x * 8) = *(const uint4*)(src + threadIdx.x * 8);
    *(uint4*)(dst + (threadIdx.x + 256) * 8) =
        *(const uint4*)(src + (threadIdx.x + 256) * 8);
#endif
  };

  stageB(0, 0);
  for (int S = 0; S < KS; ++S) {
    if (S + 1 < KS) {
      stageB(S + 1, (S + 1) & 1);           // prefetch next slab into other buf
#if USE_ASYNC_COPY
      __builtin_amdgcn_s_wait_asynccnt(2);  // slab S done; S+1 may be in flight
#endif
    } else {
#if USE_ASYNC_COPY
      __builtin_amdgcn_s_wait_asynccnt(0);
#endif
    }
    __syncthreads();

    // A fragment: contiguous 32B per lane, lane-coalesced (2x global_load_b128)
    const unsigned short* ap = Asw + ((T * KS + S) * 32 + lane) * 16;
    v16bf af = *(const v16bf*)ap;
    if (S + 1 < KS)
      __builtin_prefetch(ap + 32 * 16, 0, 1);   // next A slab -> global_prefetch

    const unsigned short* bb = &sB[S & 1][0];
#pragma unroll
    for (int t = 0; t < NT; ++t) {
      v16bf bfrag = *(const v16bf*)(bb + (t * 32 + lane) * 16);  // 2x ds_load_b128
      acc[t] = __builtin_amdgcn_wmma_f32_16x16x32_bf16(
          false, af, false, bfrag, (short)0, acc[t], false, false);
    }
    __syncthreads();
  }

  // D layout: VGPR r -> row r (lanes 0-15) / 8+r (lanes 16-31); col = lane&15
  const long cm = T * 16 + ((lane < 16) ? 0 : 8);
  const int cn = lane & 15;
#pragma unroll
  for (int t = 0; t < NT; ++t) {
    const int n = (t << 4) + cn;
    const float bv = bias ? bias[n] : 0.0f;
#pragma unroll
    for (int r = 0; r < 8; ++r) {
      float v = acc[t][r] + bv;
      if (ACT == 1) v = silu_f(v);
      C[(cm + r) * N + n] = v;
    }
  }
}

// ------------------------- swizzling conversions ---------------------------
// fp32 activations [rows, K] -> bf16 A-fragment order.
__global__ void f32_to_bf16_swzA(const float* __restrict__ x,
                                 unsigned short* __restrict__ Asw,
                                 long rows, int K)
{
  long i = (long)blockIdx.x * blockDim.x + threadIdx.x;
  long total = rows * K;
  if (i >= total) return;
  int k = (int)(i % K);
  long row = i / K;
  long T = row >> 4;
  int m16 = (int)(row & 15);
  int S = k >> 5, kk = k & 31;
  int lane = m16 + (((kk >> 3) & 1) << 4);
  int j = (kk & 7) + ((kk & 16) ? 8 : 0);
  int KS = K >> 5;
  Asw[((T * KS + S) * 32 + lane) * 16 + j] = f2bfu(x[i]);
}

// im2col for 4x4/stride-4 patch embed fused with A-swizzle (K = 512).
__global__ void im2col_patch_swzA(const float* __restrict__ f,
                                  unsigned short* __restrict__ Asw)
{
  long i = (long)blockIdx.x * blockDim.x + threadIdx.x;
  if (i >= (long)ROWS * 512) return;
  int k = (int)(i & 511);
  long row = i >> 9;
  int b = (int)(row >> 12);
  int tok = (int)(row & 4095);
  int py = tok >> 6, px = tok & 63;
  int c = k >> 4, kh = (k >> 2) & 3, kw = k & 3;
  int y = py * 4 + kh, x = px * 4 + kw;
  float v = f[((long)(b * 32 + c) * 256 + y) * 256 + x];
  long T = row >> 4;
  int m16 = (int)(row & 15);
  int S = k >> 5, kk = k & 31;
  int lane = m16 + (((kk >> 3) & 1) << 4);
  int j = (kk & 7) + ((kk & 16) ? 8 : 0);
  Asw[((T * 16 + S) * 32 + lane) * 16 + j] = f2bfu(v);   // KS = 512/32 = 16
}

// W[N, K] fp32 -> bf16 B-fragment order, slab-major, zero-padded cols.
__global__ void wt_to_bf16T_swzB(const float* __restrict__ W,
                                 unsigned short* __restrict__ Bsw,
                                 int N, int K, int Npad)
{
  int NT = Npad >> 4;
  long i = (long)blockIdx.x * blockDim.x + threadIdx.x;
  long total = (long)(K >> 5) * BSLAB;
  if (i >= total) return;
  int r = (int)(i & (BSLAB - 1));
  int S = (int)(i >> 12);
  if (r >= NT * 512) return;                 // padded tail never read by WMMA
  int t = r >> 9, lane = (r >> 4) & 31, j = r & 15;
  int k = S * 32 + ((lane >= 16) ? 8 : 0) + (j & 7) + ((j & 8) ? 16 : 0);
  int n = t * 16 + (lane & 15);
  float v = (n < N) ? W[(long)n * K + k] : 0.0f;
  Bsw[i] = f2bfu(v);
}

// ---------------------------------- convs ----------------------------------
__global__ void conv3x3_lrelu(const float* __restrict__ in,
                              const float* __restrict__ w,
                              const float* __restrict__ bias,
                              float* __restrict__ out,
                              int Bn, int Cin, int Cout, int H, int W)
{
  long i = (long)blockIdx.x * blockDim.x + threadIdx.x;
  long total = (long)Bn * Cout * H * W;
  if (i >= total) return;
  int x = (int)(i % W); long t = i / W;
  int y = (int)(t % H); t /= H;
  int oc = (int)(t % Cout); int b = (int)(t / Cout);
  float acc = bias[oc];
  for (int ic = 0; ic < Cin; ++ic) {
    const float* ip = in + (long)(b * Cin + ic) * H * W;
    const float* wp = w + (long)(oc * Cin + ic) * 9;
#pragma unroll
    for (int ky = 0; ky < 3; ++ky) {
      int yy = y + ky - 1;
      if (yy < 0 || yy >= H) continue;
#pragma unroll
      for (int kx = 0; kx < 3; ++kx) {
        int xx = x + kx - 1;
        if (xx < 0 || xx >= W) continue;
        acc += wp[ky * 3 + kx] * ip[(long)yy * W + xx];
      }
    }
  }
  out[i] = (acc > 0.0f) ? acc : 0.01f * acc;
}

// ------------------------------ SSM pieces ---------------------------------
__global__ void __launch_bounds__(128) layernorm_96(
    const float* __restrict__ x, const float* __restrict__ g,
    const float* __restrict__ b, float* __restrict__ y, int rows)
{
  int row = blockIdx.x * 4 + (threadIdx.x >> 5);
  int lane = threadIdx.x & 31;
  if (row >= rows) return;
  const float* xr = x + (long)row * DM;
  float v0 = xr[lane], v1 = xr[lane + 32], v2 = xr[lane + 64];
  float s = v0 + v1 + v2;
#pragma unroll
  for (int o = 16; o > 0; o >>= 1) s += __shfl_xor(s, o, 32);
  float mean = s * (1.0f / 96.0f);
  float d0 = v0 - mean, d1 = v1 - mean, d2 = v2 - mean;
  float q = d0 * d0 + d1 * d1 + d2 * d2;
#pragma unroll
  for (int o = 16; o > 0; o >>= 1) q += __shfl_xor(q, o, 32);
  float inv = rsqrtf(q * (1.0f / 96.0f) + 1e-5f);
  float* yr = y + (long)row * DM;
  yr[lane]      = d0 * inv * g[lane]      + b[lane];
  yr[lane + 32] = d1 * inv * g[lane + 32] + b[lane + 32];
  yr[lane + 64] = d2 * inv * g[lane + 64] + b[lane + 64];
}

__global__ void dwconv_silu(const float* __restrict__ x, const float* __restrict__ w,
                            const float* __restrict__ cb, float* __restrict__ y,
                            int Bn, int L)
{
  long i = (long)blockIdx.x * blockDim.x + threadIdx.x;
  long total = (long)Bn * L * DM;
  if (i >= total) return;
  int d = (int)(i % DM);
  long bl = i / DM;
  int l = (int)(bl % L);
  long brow0 = bl - l;                       // b*L
  float acc = cb[d];
#pragma unroll
  for (int j = 0; j < 4; ++j) {
    int ls = l - 3 + j;                      // causal pad K-1
    if (ls >= 0) acc += w[d * 4 + j] * x[(brow0 + ls) * DM + d];
  }
  y[i] = silu_f(acc);
}

__global__ void delta_softplus(const float* __restrict__ xdbl,
                               const float* __restrict__ dw,
                               const float* __restrict__ db,
                               float* __restrict__ out, int rows)
{
  long i = (long)blockIdx.x * blockDim.x + threadIdx.x;
  long total = (long)rows * DM;
  if (i >= total) return;
  int d = (int)(i % DM);
  long r = i / DM;
  float acc = db[d];
#pragma unroll
  for (int k = 0; k < 6; ++k) acc += xdbl[r * NPAD + k] * dw[d * 6 + k];
  out[i] = (acc > 20.0f) ? acc : log1pf(__expf(acc));
}

__global__ void __launch_bounds__(128) row_dot_96(
    const float* __restrict__ a, const float* __restrict__ b,
    float* __restrict__ o, int rows)
{
  int row = blockIdx.x * 4 + (threadIdx.x >> 5);
  int lane = threadIdx.x & 31;
  if (row >= rows) return;
  const float* ar = a + (long)row * DM;
  const float* br = b + (long)row * DM;
  float s = ar[lane] * br[lane] + ar[lane + 32] * br[lane + 32]
          + ar[lane + 64] * br[lane + 64];
#pragma unroll
  for (int off = 16; off > 0; off >>= 1) s += __shfl_xor(s, off, 32);
  if (lane == 0) o[row] = s;
}

__global__ void aneg_k(const float* __restrict__ a, float* __restrict__ o, int n) {
  int i = blockIdx.x * blockDim.x + threadIdx.x;
  if (i < n) o[i] = -__expf(a[i]);
}

// sequential selective scan: one block per batch, one thread per channel d,
// h[16] state resident in VGPRs, B/C broadcast through LDS each step.
__global__ void __launch_bounds__(96) selective_scan(
    const float* __restrict__ delta, const float* __restrict__ xdbl,
    const float* __restrict__ s, const float* __restrict__ An,
    const float* __restrict__ x, const float* __restrict__ Dp,
    float* __restrict__ y, int L)
{
  __shared__ float sBC[32];
  int b = blockIdx.x;
  int d = threadIdx.x;
  float h[DSTATE], Ad[DSTATE];
#pragma unroll
  for (int n = 0; n < DSTATE; ++n) { h[n] = 0.0f; Ad[n] = An[d * DSTATE + n]; }
  float Dd = Dp[d];
  for (int l = 0; l < L; ++l) {
    long row = (long)b * L + l;
    if (threadIdx.x < 32) sBC[threadIdx.x] = xdbl[row * NPAD + 6 + threadIdx.x];
    __syncthreads();
    float dlt = delta[row * DM + d];
    float sv  = s[row];
    float acc = 0.0f;
#pragma unroll
    for (int n = 0; n < DSTATE; ++n) {
      h[n] = dlt * Ad[n] * h[n] + sv * sBC[n];
      acc += h[n] * sBC[16 + n];
    }
    y[row * DM + d] = acc + x[row * DM + d] * Dd;
    __syncthreads();
  }
}

// ----------------------------- elementwise ---------------------------------
__global__ void ewise_exchange(const float* __restrict__ t1, const float* __restrict__ t2,
                               float* __restrict__ e1, float* __restrict__ e2, long n) {
  long i = (long)blockIdx.x * blockDim.x + threadIdx.x;
  if (i >= n) return;
  bool lo = (int)(i % DM) < (DM / 2);
  e1[i] = lo ? t1[i] : t2[i];
  e2[i] = lo ? t2[i] : t1[i];
}
__global__ void ewise_avg(const float* __restrict__ a, const float* __restrict__ b,
                          float* __restrict__ o, long n) {
  long i = (long)blockIdx.x * blockDim.x + threadIdx.x;
  if (i < n) o[i] = (a[i] + b[i]) * 0.5f;
}
__global__ void ewise_gatesum(const float* __restrict__ a, const float* __restrict__ b,
                              const float* __restrict__ z, float* __restrict__ o, long n) {
  long i = (long)blockIdx.x * blockDim.x + threadIdx.x;
  if (i < n) o[i] = (a[i] + b[i]) * z[i];
}
__global__ void ewise_add(float* __restrict__ dst, const float* __restrict__ src, long n) {
  long i = (long)blockIdx.x * blockDim.x + threadIdx.x;
  if (i < n) dst[i] += src[i];
}

// ------------------ patch unembed + bilinear x4 + 1x1 conv -----------------
__global__ void unembed_resize_conv(const float* __restrict__ tok,
                                    const float* __restrict__ puw,
                                    const float* __restrict__ pub,
                                    float* __restrict__ out)
{
  long i = (long)blockIdx.x * blockDim.x + threadIdx.x;
  long total = (long)BATCH * 256 * 256;
  if (i >= total) return;
  int x = (int)(i & 255);
  long t = i >> 8;
  int y = (int)(t & 255);
  int b = (int)(t >> 8);
  float fy = (y + 0.5f) * 0.25f - 0.5f;
  float fx = (x + 0.5f) * 0.25f - 0.5f;
  int iy = (int)floorf(fy), ix = (int)floorf(fx);
  float wy = fy - (float)iy, wx = fx - (float)ix;
  int y0c = min(63, max(0, iy)),     y1c = min(63, max(0, iy + 1));
  int x0c = min(63, max(0, ix)),     x1c = min(63, max(0, ix + 1));
  const float* T = tok + (long)b * LTOK * DM;
  long r00 = ((long)y0c * 64 + x0c) * DM;
  long r01 = ((long)y0c * 64 + x1c) * DM;
  long r10 = ((long)y1c * 64 + x0c) * DM;
  long r11 = ((long)y1c * 64 + x1c) * DM;
  float w00 = (1 - wy) * (1 - wx), w01 = (1 - wy) * wx;
  float w10 = wy * (1 - wx),       w11 = wy * wx;
  float a0 = pub[0], a1 = pub[1], a2 = pub[2];
  for (int c = 0; c < DM; ++c) {
    float v = w00 * T[r00 + c] + w01 * T[r01 + c]
            + w10 * T[r10 + c] + w11 * T[r11 + c];
    a0 += puw[c] * v;
    a1 += puw[DM + c] * v;
    a2 += puw[2 * DM + c] * v;
  }
  long o = ((long)b * 3 * 256 + y) * 256 + x;
  out[o]          = a0;
  out[o + 65536]  = a1;
  out[o + 131072] = a2;
}

// ---------------------------------------------------------------------------
// Host orchestration.  Input leaf order (jax pytree, alphabetical dict keys):
//   0:i1 1:i2 | 2..45 enc1[0..3]x{A_log,D,cb,cw,db,dw,nb,ng,pw,xb,xw}
//   46..89 enc2 | 90 fb 91 fw | 92..99 l1c1b,l1c1w,l1c2b,l1c2w,l2c1b,l2c1w,
//   l2c2b,l2c2w | 100..110 lfl1 | 111..121 lfl2 | 122..132 mmb_b1 |
//   133..143 mmb_b2 | 144 pe1b 145 pe1w 146 pe2b 147 pe2w | 148 pub 149 puw |
//   150 zb 151 zw
// ---------------------------------------------------------------------------
extern "C" void kernel_launch(void* const* d_in, const int* in_sizes, int n_in,
                              void* d_out, int out_size, void* d_ws, size_t ws_size,
                              hipStream_t stream)
{
  (void)in_sizes; (void)n_in; (void)out_size; (void)ws_size;
  auto F = [&](int i) { return (const float*)d_in[i]; };

  char* base = (char*)d_ws;
  size_t off = 0;
  auto alloc = [&](size_t bytes) -> void* {
    off = (off + 255) & ~(size_t)255;
    void* p = base + off;
    off += bytes;
    return p;
  };

  const size_t CONV_B = (size_t)BATCH * 32 * 256 * 256 * 4;
  const size_t TOK_B  = (size_t)ROWS * DM * 4;
  const long   CONVN  = (long)BATCH * 32 * 256 * 256;
  const long   NTOK   = (long)ROWS * DM;

  float* ctmp = (float*)alloc(CONV_B);
  float* f1b  = (float*)alloc(CONV_B);
  float* f2b  = (float*)alloc(CONV_B);
  float* t1   = (float*)alloc(TOK_B);
  float* t2   = (float*)alloc(TOK_B);
  float* tsw  = (float*)alloc(TOK_B);
  float* bLN  = (float*)alloc(TOK_B);
  float* bX   = (float*)alloc(TOK_B);
  float* bXC  = (float*)alloc(TOK_B);
  float* bDEL = (float*)alloc(TOK_B);
  float* e1   = (float*)alloc(TOK_B);
  float* e2   = (float*)alloc(TOK_B);
  float* ffb  = (float*)alloc(TOK_B);
  float* yb1  = (float*)alloc(TOK_B);
  float* yb2  = (float*)alloc(TOK_B);
  float* zbuf = (float*)alloc(TOK_B);
  float* bXD  = (float*)alloc((size_t)ROWS * NPAD * 4);
  float* bS   = (float*)alloc((size_t)ROWS * 4);
  float* bAN  = (float*)alloc((size_t)DM * DSTATE * 4);
  unsigned short* A16 = (unsigned short*)alloc((size_t)ROWS * 512 * 2);
  unsigned short* W16 = (unsigned short*)alloc((size_t)16 * BSLAB * 2); // 16 slabs max

  auto g1 = [](long n) { return (unsigned)((n + 255) / 256); };

  // ---- local feature extraction (4 direct convs, leaky-relu) ----
  conv3x3_lrelu<<<g1(CONVN), 256, 0, stream>>>(F(0), F(93), F(92), ctmp, BATCH, 3, 32, 256, 256);
  conv3x3_lrelu<<<g1(CONVN), 256, 0, stream>>>(ctmp, F(95), F(94), f1b,  BATCH, 32, 32, 256, 256);
  conv3x3_lrelu<<<g1(CONVN), 256, 0, stream>>>(F(1), F(97), F(96), ctmp, BATCH, 1, 32, 256, 256);
  conv3x3_lrelu<<<g1(CONVN), 256, 0, stream>>>(ctmp, F(99), F(98), f2b,  BATCH, 32, 32, 256, 256);

  // ---- patch embed: fused im2col+swizzle, then WMMA GEMM (K=512, N=96) ----
  im2col_patch_swzA<<<g1((long)ROWS * 512), 256, 0, stream>>>(f1b, A16);
  wt_to_bf16T_swzB<<<g1(16L * BSLAB), 256, 0, stream>>>(F(145), W16, 96, 512, 96);
  gemm_bf16_wmma<6, 0><<<ROWS / 128, 256, 0, stream>>>(A16, W16, F(144), t1, 512);
  im2col_patch_swzA<<<g1((long)ROWS * 512), 256, 0, stream>>>(f2b, A16);
  wt_to_bf16T_swzB<<<g1(16L * BSLAB), 256, 0, stream>>>(F(147), W16, 96, 512, 96);
  gemm_bf16_wmma<6, 0><<<ROWS / 128, 256, 0, stream>>>(A16, W16, F(146), t2, 512);

  // ---- one SSM branch (Mamba) ----
  auto ssm_branch = [&](const float* in, float* out, int L0) {
    const float* A_log = F(L0 + 0); const float* Dp = F(L0 + 1);
    const float* cb = F(L0 + 2);    const float* cw = F(L0 + 3);
    const float* db = F(L0 + 4);    const float* dw = F(L0 + 5);
    const float* nb = F(L0 + 6);    const float* ng = F(L0 + 7);
    const float* pw = F(L0 + 8);    const float* xb = F(L0 + 9);
    const float* xw = F(L0 + 10);
    layernorm_96<<<ROWS / 4, 128, 0, stream>>>(in, ng, nb, bLN, ROWS);
    wt_to_bf16T_swzB<<<g1(3L * BSLAB), 256, 0, stream>>>(xw, W16, 96, 96, 96);
    f32_to_bf16_swzA<<<g1(NTOK), 256, 0, stream>>>(bLN, A16, ROWS, 96);
    gemm_bf16_wmma<6, 0><<<ROWS / 128, 256, 0, stream>>>(A16, W16, xb, bX, 96);
    dwconv_silu<<<g1(NTOK), 256, 0, stream>>>(bX, cw, cb, bXC, BATCH, LTOK);
    wt_to_bf16T_swzB<<<g1(3L * BSLAB), 256, 0, stream>>>(pw, W16, 38, 96, NPAD);
    f32_to_bf16_swzA<<<g1(NTOK), 256, 0, stream>>>(bXC, A16, ROWS, 96);
    gemm_bf16_wmma<3, 0><<<ROWS / 128, 256, 0, stream>>>(A16, W16, nullptr, bXD, 96);
    delta_softplus<<<g1(NTOK), 256, 0, stream>>>(bXD, dw, db, bDEL, ROWS);
    row_dot_96<<<ROWS / 4, 128, 0, stream>>>(bDEL, bXC, bS, ROWS);
    aneg_k<<<g1(DM * DSTATE), 256, 0, stream>>>(A_log, bAN, DM * DSTATE);
    selective_scan<<<BATCH, 96, 0, stream>>>(bDEL, bXD, bS, bAN, bXC, Dp, out, LTOK);
  };

  // ---- modality encoders (4 blocks each, ping-pong buffers) ----
  float* cur1 = t1; float* alt = tsw;
  for (int k = 0; k < 4; ++k) {
    ssm_branch(cur1, alt, 2 + 11 * k);
    float* tt = cur1; cur1 = alt; alt = tt;          // ends back at t1
  }
  float* cur2 = t2; alt = tsw;
  for (int k = 0; k < 4; ++k) {
    ssm_branch(cur2, alt, 46 + 11 * k);
    float* tt = cur2; cur2 = alt; alt = tt;          // ends back at t2
  }

  // ---- channel exchange + low-level fusion layers ----
  ewise_exchange<<<g1(NTOK), 256, 0, stream>>>(cur1, cur2, e1, e2, NTOK);
  ssm_branch(e1, t1, 100);   // lfl1
  ssm_branch(e2, t2, 111);   // lfl2
  ewise_avg<<<g1(NTOK), 256, 0, stream>>>(t1, t2, ffb, NTOK);

  // ---- MMB ----
  ssm_branch(t1, yb1, 122);  // mmb_b1
  ssm_branch(t2, yb2, 133);  // mmb_b2
  // z = silu(ff @ zw^T + zb)   (SiLU fused into GEMM epilogue)
  wt_to_bf16T_swzB<<<g1(3L * BSLAB), 256, 0, stream>>>(F(151), W16, 96, 96, 96);
  f32_to_bf16_swzA<<<g1(NTOK), 256, 0, stream>>>(ffb, A16, ROWS, 96);
  gemm_bf16_wmma<6, 1><<<ROWS / 128, 256, 0, stream>>>(A16, W16, F(150), zbuf, 96);
  // (y1+y2)*z -> fw linear -> + ff
  ewise_gatesum<<<g1(NTOK), 256, 0, stream>>>(yb1, yb2, zbuf, e1, NTOK);
  wt_to_bf16T_swzB<<<g1(3L * BSLAB), 256, 0, stream>>>(F(91), W16, 96, 96, 96);
  f32_to_bf16_swzA<<<g1(NTOK), 256, 0, stream>>>(e1, A16, ROWS, 96);
  gemm_bf16_wmma<6, 0><<<ROWS / 128, 256, 0, stream>>>(A16, W16, F(90), e2, 96);
  ewise_add<<<g1(NTOK), 256, 0, stream>>>(e2, ffb, NTOK);

  // ---- patch unembed + bilinear x4 + 1x1 conv ----
  unembed_resize_conv<<<g1((long)BATCH * 256 * 256), 256, 0, stream>>>(
      e2, F(149), F(148), (float*)d_out);
}
